// VisionMambaLayer_25795573580035
// MI455X (gfx1250) — matrix operand order repaired and checked by reading, not compiled, not run on hardware
//
#include <hip/hip_runtime.h>
#include <hip/hip_bf16.h>

// ---------------------------------------------------------------------------
// VisionMamba layer for MI455X (gfx1250): bf16 WMMA GEMMs + Prim MST + tree scan
// GEMM uses 2x4 register blocking per wave: 8 v_wmma per k-step, A-frag reuse.
// ---------------------------------------------------------------------------

typedef __bf16 bf16;
typedef __attribute__((ext_vector_type(16))) __bf16 v16bf;
typedef __attribute__((ext_vector_type(8)))  __bf16 v8bf;
typedef __attribute__((ext_vector_type(8)))  float   v8f;

#define BB   16
#define HH   48
#define WW   48
#define CC   256
#define NN   2304          // HH*WW
#define DI   512           // D_INNER
#define MM   36864         // BB*NN
#define HID  1024
#define DTR  16            // DT_RANK

// ------------------------------ helpers ------------------------------------

__device__ __forceinline__ float siluf(float v)  { return v / (1.f + __expf(-v)); }
__device__ __forceinline__ float geluf(float v)  { return 0.5f * v * (1.f + erff(v * 0.70710678118654752f)); }
__device__ __forceinline__ float softplusf(float v) { return (v > 20.f) ? v : log1pf(expf(v)); }

// ------------------------- weight transpose+convert -------------------------
// src: fp32 [K, Ncol] row-major  ->  dst: bf16 [Ncol, K] row-major
__global__ __launch_bounds__(256)
void tconv_kernel(const float* __restrict__ src, bf16* __restrict__ dst, int K, int Ncol) {
    int idx = blockIdx.x * 256 + threadIdx.x;
    if (idx >= K * Ncol) return;
    int n = idx / K, k = idx - n * K;
    dst[idx] = (bf16)src[(size_t)k * Ncol + n];
}

// ------------------------------ layernorm -----------------------------------
// one wave per row of 256; writes bf16 normalized row
__global__ __launch_bounds__(256)
void ln_kernel(const float* __restrict__ x, const float* __restrict__ g,
               const float* __restrict__ b, bf16* __restrict__ out, int rows) {
    int wave = threadIdx.x >> 5, lane = threadIdx.x & 31;
    int row = blockIdx.x * 8 + wave;
    if (row >= rows) return;
    const float* xr = x + (size_t)row * CC;
    float v[8]; float s = 0.f, sq = 0.f;
#pragma unroll
    for (int i = 0; i < 8; ++i) { float t = xr[lane + 32 * i]; v[i] = t; s += t; sq += t * t; }
#pragma unroll
    for (int m = 16; m >= 1; m >>= 1) { s += __shfl_xor(s, m, 32); sq += __shfl_xor(sq, m, 32); }
    float mean = s * (1.f / 256.f);
    float var  = sq * (1.f / 256.f) - mean * mean;
    float rstd = rsqrtf(var + 1e-5f);
    bf16* orow = out + (size_t)row * CC;
#pragma unroll
    for (int i = 0; i < 8; ++i) {
        int k = lane + 32 * i;
        orow[k] = (bf16)((v[i] - mean) * rstd * g[k] + b[k]);
    }
}

// --------------------------- WMMA GEMM (bf16 -> f32) ------------------------
// A: bf16 [M,K] row-major.  Wt: bf16 [Ncol,K] row-major (weight transposed).
// Wave: 2(M) x 4(N) 16x16 tiles -> 8 WMMA per 32-K step, A frags reused 4x.
// Block: 8 waves = 4(M) x 2(N) wave grid -> 128 x 128 output tile.
__device__ __forceinline__ v16bf frag_from_chunks(const bf16* p0, const bf16* p1) {
    v8bf a0 = *(const v8bf*)p0;
    v8bf a1 = *(const v8bf*)p1;
    v16bf f;
#pragma unroll
    for (int i = 0; i < 8; ++i) { f[i] = a0[i]; f[8 + i] = a1[i]; }
    return f;
}

template<int EPI>
__global__ __launch_bounds__(256)
void gemm_wmma_kernel(const bf16* __restrict__ A, const bf16* __restrict__ Wt,
                      int K, int Ncol,
                      const float* __restrict__ bias,
                      const float* __restrict__ resid,
                      float* __restrict__ outF0, float* __restrict__ outF1,
                      bf16*  __restrict__ outB) {
    const int wave = threadIdx.x >> 5;
    const int lane = threadIdx.x & 31;
    const int half = lane >> 4;     // K-half selector: contiguous k = 16*half + s for A and B
    const int l    = lane & 15;
    const int tileM0 = blockIdx.y * 128 + (wave >> 1) * 32;  // two 16-row tiles
    const int tileN0 = blockIdx.x * 128 + (wave & 1) * 64;   // four 16-col tiles

    const bf16* arow0 = A  + (size_t)(tileM0 + l) * K;
    const bf16* arow1 = A  + (size_t)(tileM0 + 16 + l) * K;
    const bf16* brow  = Wt + (size_t)(tileN0 + l) * K;
    const size_t bstride = (size_t)16 * K;                   // next 16-col tile

    v8f acc[2][4] = {};
    for (int k0 = 0; k0 < K; k0 += 32) {
        // A 16x32: lane<16 -> K 0..15; lane>=16 -> K 16..31
        v16bf a0 = frag_from_chunks(arow0 + k0 + half * 16, arow0 + k0 + half * 16 + 8);
        v16bf a1 = frag_from_chunks(arow1 + k0 + half * 16, arow1 + k0 + half * 16 + 8);
        // B 32x16 (from transposed weight row): lane<16 -> K 0..15; lane>=16 -> K 16..31
        v16bf bf0 = frag_from_chunks(brow + 0 * bstride + k0 + half * 16,
                                     brow + 0 * bstride + k0 + half * 16 + 8);
        v16bf bf1 = frag_from_chunks(brow + 1 * bstride + k0 + half * 16,
                                     brow + 1 * bstride + k0 + half * 16 + 8);
        v16bf bf2 = frag_from_chunks(brow + 2 * bstride + k0 + half * 16,
                                     brow + 2 * bstride + k0 + half * 16 + 8);
        v16bf bf3 = frag_from_chunks(brow + 3 * bstride + k0 + half * 16,
                                     brow + 3 * bstride + k0 + half * 16 + 8);
        acc[0][0] = __builtin_amdgcn_wmma_f32_16x16x32_bf16(false, a0, false, bf0, (short)0, acc[0][0], false, false);
        acc[0][1] = __builtin_amdgcn_wmma_f32_16x16x32_bf16(false, a0, false, bf1, (short)0, acc[0][1], false, false);
        acc[0][2] = __builtin_amdgcn_wmma_f32_16x16x32_bf16(false, a0, false, bf2, (short)0, acc[0][2], false, false);
        acc[0][3] = __builtin_amdgcn_wmma_f32_16x16x32_bf16(false, a0, false, bf3, (short)0, acc[0][3], false, false);
        acc[1][0] = __builtin_amdgcn_wmma_f32_16x16x32_bf16(false, a1, false, bf0, (short)0, acc[1][0], false, false);
        acc[1][1] = __builtin_amdgcn_wmma_f32_16x16x32_bf16(false, a1, false, bf1, (short)0, acc[1][1], false, false);
        acc[1][2] = __builtin_amdgcn_wmma_f32_16x16x32_bf16(false, a1, false, bf2, (short)0, acc[1][2], false, false);
        acc[1][3] = __builtin_amdgcn_wmma_f32_16x16x32_bf16(false, a1, false, bf3, (short)0, acc[1][3], false, false);
    }

#pragma unroll
    for (int i = 0; i < 2; ++i) {
#pragma unroll
        for (int j = 0; j < 4; ++j) {
            const int col = tileN0 + j * 16 + l;
            const float bcol = bias[col];
#pragma unroll
            for (int r = 0; r < 8; ++r) {
                int row = tileM0 + i * 16 + r + half * 8;
                float v = acc[i][j][r] + bcol;
                if constexpr (EPI == 0) {
                    // split: cols [0,512) -> silu -> xi ; cols [512,1024) -> z
                    if (col < DI) outF0[(size_t)row * DI + col] = siluf(v);
                    else          outF1[(size_t)row * DI + (col - DI)] = v;
                } else if constexpr (EPI == 1) {
                    // + residual -> fp32
                    outF0[(size_t)row * Ncol + col] = v + resid[(size_t)row * Ncol + col];
                } else if constexpr (EPI == 2) {
                    // gelu -> bf16
                    outB[(size_t)row * Ncol + col] = (bf16)geluf(v);
                } else { // EPI == 3: + residual -> final output fp32
                    outF0[(size_t)row * Ncol + col] = v + resid[(size_t)row * Ncol + col];
                }
            }
        }
    }
}

// ----------------------- MST: node norms + edge weights ---------------------
__global__ __launch_bounds__(256)
void norms_kernel(const float* __restrict__ x, float* __restrict__ nrm, int total) {
    int wave = threadIdx.x >> 5, lane = threadIdx.x & 31;
    int node = blockIdx.x * 8 + wave;
    if (node >= total) return;
    const float* xr = x + (size_t)node * CC;
    float s = 0.f;
#pragma unroll
    for (int i = 0; i < 8; ++i) { float t = xr[lane + 32 * i]; s += t * t; }
#pragma unroll
    for (int m = 16; m >= 1; m >>= 1) s += __shfl_xor(s, m, 32);
    if (lane == 0) nrm[node] = sqrtf(s) + 1e-8f;
}

__global__ __launch_bounds__(256)
void edges_kernel(const float* __restrict__ x, const float* __restrict__ nrm,
                  float* __restrict__ wgt) {
    int wave = threadIdx.x >> 5, lane = threadIdx.x & 31;
    int g = blockIdx.x * 8 + wave;          // global node id in [0, B*N)
    if (g >= BB * NN) return;
    int b = g / NN, n = g - b * NN;
    int r = n / WW, c = n - r * WW;
    int nb[4];
    nb[0] = (r > 0)      ? n - WW : -1;   // up
    nb[1] = (r < HH - 1) ? n + WW : -1;   // down
    nb[2] = (c > 0)      ? n - 1  : -1;   // left
    nb[3] = (c < WW - 1) ? n + 1  : -1;   // right
    const float* xb = x + (size_t)b * NN * CC;
    const float* xn = xb + (size_t)n * CC;
    float d[4] = {0.f, 0.f, 0.f, 0.f};
#pragma unroll
    for (int i = 0; i < 8; ++i) {
        int k = lane + 32 * i;
        float t = xn[k];
#pragma unroll
        for (int j = 0; j < 4; ++j)
            if (nb[j] >= 0) d[j] += t * xb[(size_t)nb[j] * CC + k];
    }
#pragma unroll
    for (int j = 0; j < 4; ++j)
#pragma unroll
        for (int m = 16; m >= 1; m >>= 1) d[j] += __shfl_xor(d[j], m, 32);
    if (lane < 4) {
        int j = lane;
        float w;
        if (nb[j] >= 0) {
            float cosv = d[j] / (nrm[(size_t)b * NN + n] * nrm[(size_t)b * NN + nb[j]]);
            w = expf(1.f - cosv);
        } else {
            w = __builtin_inff();
        }
        wgt[((size_t)b * NN + n) * 4 + j] = w;
    }
}

// ------------------------------ Prim MST ------------------------------------
__global__ __launch_bounds__(256)
void mst_kernel(const float* __restrict__ wgt, int* __restrict__ order,
                int* __restrict__ par) {
    __shared__ float key[NN];
    __shared__ int   spar[NN];
    __shared__ unsigned char intree[NN];
    __shared__ unsigned long long red[8];
    __shared__ unsigned long long bestSh;

    const int b = blockIdx.x, tid = threadIdx.x;
    const int wave = tid >> 5, lane = tid & 31;
    for (int i = tid; i < NN; i += 256) {
        key[i] = (i == 0) ? 0.f : __builtin_inff();
        spar[i] = 0;
        intree[i] = 0;
    }
    __syncthreads();
    const float* wb = wgt + (size_t)b * NN * 4;

    for (int t = 0; t < NN; ++t) {
        // argmin over non-tree keys; pack (fbits<<32)|idx: all weights >= 0 so
        // uint bit order == float order, and ties pick the smallest index.
        unsigned long long m = ~0ull;
        for (int i = tid; i < NN; i += 256) {
            if (!intree[i]) {
                unsigned long long pk =
                    ((unsigned long long)__float_as_uint(key[i]) << 32) | (unsigned int)i;
                if (pk < m) m = pk;
            }
        }
#pragma unroll
        for (int s = 16; s >= 1; s >>= 1) {
            unsigned long long o = __shfl_xor(m, s, 32);
            if (o < m) m = o;
        }
        if (lane == 0) red[wave] = m;
        __syncthreads();
        if (tid == 0) {
            unsigned long long mm = red[0];
#pragma unroll
            for (int i = 1; i < 8; ++i) if (red[i] < mm) mm = red[i];
            bestSh = mm;
        }
        __syncthreads();
        const int u = (int)(bestSh & 0xffffffffu);
        if (tid == 0) { intree[u] = 1; order[(size_t)b * NN + t] = u; }
        if (tid < 4) {
            int r = u / WW, c = u - (u / WW) * WW;
            int nb = -1;
            if (tid == 0 && r > 0)      nb = u - WW;
            if (tid == 1 && r < HH - 1) nb = u + WW;
            if (tid == 2 && c > 0)      nb = u - 1;
            if (tid == 3 && c < WW - 1) nb = u + 1;
            if (nb >= 0 && !intree[nb]) {
                float wv = wb[(size_t)u * 4 + tid];
                if (wv < key[nb]) { key[nb] = wv; spar[nb] = u; }
            }
        }
        __syncthreads();
    }
    for (int i = tid; i < NN; i += 256) par[(size_t)b * NN + i] = spar[i];
}

// ------------------- xi @ W_xp  (K=512 -> 18 cols) --------------------------
__global__ __launch_bounds__(256)
void dbc_kernel(const float* __restrict__ xi, const float* __restrict__ Wxp,
                float* __restrict__ dbc, int rows) {
    int wave = threadIdx.x >> 5, lane = threadIdx.x & 31;
    int row = blockIdx.x * 8 + wave;
    if (row >= rows) return;
    const float* xr = xi + (size_t)row * DI;
    float acc[18];
#pragma unroll
    for (int j = 0; j < 18; ++j) acc[j] = 0.f;
    for (int k = lane; k < DI; k += 32) {
        float t = xr[k];
        const float* wr = Wxp + (size_t)k * 18;
#pragma unroll
        for (int j = 0; j < 18; ++j) acc[j] += t * wr[j];
    }
#pragma unroll
    for (int j = 0; j < 18; ++j)
#pragma unroll
        for (int m = 16; m >= 1; m >>= 1) acc[j] += __shfl_xor(acc[j], m, 32);
    if (lane == 0) {
        float* dr = dbc + (size_t)row * 18;
#pragma unroll
        for (int j = 0; j < 18; ++j) dr[j] = acc[j];
    }
}

// --------- dt = softplus(dtr@W_dt + b_dt); decay = exp(dt*A); Bx ------------
__global__ __launch_bounds__(256)
void dt_kernel(const float* __restrict__ dbc, const float* __restrict__ Wdt,
               const float* __restrict__ bdt, const float* __restrict__ Alog,
               const float* __restrict__ xi,
               float* __restrict__ decay, float* __restrict__ bx) {
    size_t i = (size_t)blockIdx.x * 256 + threadIdx.x;
    if (i >= (size_t)MM * DI) return;
    size_t row = i >> 9;
    int ch = (int)(i & (DI - 1));
    const float* dr = dbc + row * 18;
    float a = bdt[ch];
#pragma unroll
    for (int r = 0; r < DTR; ++r) a += dr[r] * Wdt[(size_t)r * DI + ch];
    float dt = softplusf(a);
    float A  = -expf(Alog[ch]);
    decay[i] = expf(dt * A);
    bx[i]    = dt * dr[16] * xi[i];   // dr[16] = Bv
}

// ------------------------- tree selective scan ------------------------------
// h stored in-place over bx: h[v] = decay[v]*h[par[v]] + Bx[v]
__global__ __launch_bounds__(512)
void scan_kernel(const int* __restrict__ order, const int* __restrict__ par,
                 const float* __restrict__ decay, float* __restrict__ h) {
    const int b = blockIdx.x, tid = threadIdx.x;
    const int* ob = order + (size_t)b * NN;
    const int* pb = par   + (size_t)b * NN;
    const size_t base = (size_t)b * NN * DI;
    for (int t = 0; t < NN; ++t) {
        int v = ob[t];
        int p = pb[v];
        float hp = (t == 0) ? 0.f : h[base + (size_t)p * DI + tid];
        size_t vi = base + (size_t)v * DI + tid;
        h[vi] = decay[vi] * hp + h[vi];
        __syncthreads();
    }
}

// -------------------- u = (Cv*h + Dp*xi) * silu(z) -> bf16 ------------------
__global__ __launch_bounds__(256)
void gate_kernel(const float* __restrict__ h, const float* __restrict__ xi,
                 const float* __restrict__ z, const float* __restrict__ dbc,
                 const float* __restrict__ Dp, bf16* __restrict__ u) {
    size_t i = (size_t)blockIdx.x * 256 + threadIdx.x;
    if (i >= (size_t)MM * DI) return;
    size_t row = i >> 9;
    int ch = (int)(i & (DI - 1));
    float Cv = dbc[row * 18 + 17];
    float y  = Cv * h[i] + Dp[ch] * xi[i];
    u[i] = (bf16)(y * siluf(z[i]));
}

// ------------------------------- launcher -----------------------------------

extern "C" void kernel_launch(void* const* d_in, const int* in_sizes, int n_in,
                              void* d_out, int out_size, void* d_ws, size_t ws_size,
                              hipStream_t stream) {
    const float* x       = (const float*)d_in[0];
    const float* n1w     = (const float*)d_in[1];
    const float* n1b     = (const float*)d_in[2];
    const float* n2w     = (const float*)d_in[3];
    const float* n2b     = (const float*)d_in[4];
    const float* W_in    = (const float*)d_in[5];
    const float* b_in    = (const float*)d_in[6];
    const float* W_xp    = (const float*)d_in[7];
    const float* W_dt    = (const float*)d_in[8];
    const float* b_dt    = (const float*)d_in[9];
    const float* A_log   = (const float*)d_in[10];
    const float* Dp      = (const float*)d_in[11];
    const float* W_out   = (const float*)d_in[12];
    const float* b_out   = (const float*)d_in[13];
    const float* fc1_w   = (const float*)d_in[14];
    const float* fc1_b   = (const float*)d_in[15];
    const float* fc2_w   = (const float*)d_in[16];
    const float* fc2_b   = (const float*)d_in[17];
    float* out = (float*)d_out;

    // ---- workspace layout (all sizes multiples of 256 bytes) ----
    char* ws = (char*)d_ws;
    size_t off = 0;
    auto alloc = [&](size_t bytes) { size_t o = off; off += (bytes + 255) & ~size_t(255); return o; };
    size_t oXNBF = alloc((size_t)MM * CC * 2);       // ln1(x) bf16
    size_t oXI   = alloc((size_t)MM * DI * 4);       // silu branch fp32
    size_t oZ    = alloc((size_t)MM * DI * 4);       // gate branch fp32
    size_t oDEC  = alloc((size_t)MM * DI * 4);       // decay fp32
    size_t oBX   = alloc((size_t)MM * DI * 4);       // Bx, becomes h in-place
    size_t oUBF  = alloc((size_t)MM * DI * 2);       // gated y bf16
    size_t oX2   = alloc((size_t)MM * CC * 4);       // x + ssm out fp32
    size_t oX2N  = alloc((size_t)MM * CC * 2);       // ln2(x2) bf16
    size_t oH1   = alloc((size_t)MM * HID * 2);      // gelu(fc1) bf16
    size_t oORD  = alloc((size_t)BB * NN * 4);
    size_t oPAR  = alloc((size_t)BB * NN * 4);
    size_t oDBC  = alloc((size_t)MM * 18 * 4);
    size_t oNRM  = alloc((size_t)BB * NN * 4);
    size_t oWGT  = alloc((size_t)BB * NN * 4 * 4);
    size_t oWTIN = alloc((size_t)1024 * 256 * 2);    // W_in^T  bf16 [1024,256]
    size_t oWTOU = alloc((size_t)256 * 512 * 2);     // W_out^T bf16 [256,512]
    size_t oWTF1 = alloc((size_t)1024 * 256 * 2);    // fc1^T   bf16 [1024,256]
    size_t oWTF2 = alloc((size_t)256 * 1024 * 2);    // fc2^T   bf16 [256,1024]
    (void)ws_size; (void)in_sizes; (void)n_in; (void)out_size;

    bf16*  xnbf  = (bf16*)(ws + oXNBF);
    float* xi    = (float*)(ws + oXI);
    float* zb    = (float*)(ws + oZ);
    float* dec   = (float*)(ws + oDEC);
    float* bx    = (float*)(ws + oBX);
    bf16*  ubf   = (bf16*)(ws + oUBF);
    float* x2    = (float*)(ws + oX2);
    bf16*  x2n   = (bf16*)(ws + oX2N);
    bf16*  h1    = (bf16*)(ws + oH1);
    int*   ordp  = (int*)(ws + oORD);
    int*   parp  = (int*)(ws + oPAR);
    float* dbc   = (float*)(ws + oDBC);
    float* nrm   = (float*)(ws + oNRM);
    float* wgt   = (float*)(ws + oWGT);
    bf16*  WtIn  = (bf16*)(ws + oWTIN);
    bf16*  WtOut = (bf16*)(ws + oWTOU);
    bf16*  WtF1  = (bf16*)(ws + oWTF1);
    bf16*  WtF2  = (bf16*)(ws + oWTF2);

    // ---- weight transposes to bf16 ----
    tconv_kernel<<<(256 * 1024 + 255) / 256, 256, 0, stream>>>(W_in,  WtIn,  256,  1024);
    tconv_kernel<<<(512 * 256  + 255) / 256, 256, 0, stream>>>(W_out, WtOut, 512,  256);
    tconv_kernel<<<(256 * 1024 + 255) / 256, 256, 0, stream>>>(fc1_w, WtF1,  256,  1024);
    tconv_kernel<<<(1024 * 256 + 255) / 256, 256, 0, stream>>>(fc2_w, WtF2,  1024, 256);

    // ---- MST path (on raw x) ----
    norms_kernel<<<MM / 8, 256, 0, stream>>>(x, nrm, MM);
    edges_kernel<<<MM / 8, 256, 0, stream>>>(x, nrm, wgt);
    mst_kernel<<<BB, 256, 0, stream>>>(wgt, ordp, parp);

    // ---- SSM branch ----
    ln_kernel<<<MM / 8, 256, 0, stream>>>(x, n1w, n1b, xnbf, MM);
    // xz = xn @ W_in + b_in, split into silu(xi) / z
    gemm_wmma_kernel<0><<<dim3(1024 / 128, MM / 128), 256, 0, stream>>>(
        xnbf, WtIn, 256, 1024, b_in, nullptr, xi, zb, nullptr);
    dbc_kernel<<<MM / 8, 256, 0, stream>>>(xi, W_xp, dbc, MM);
    dt_kernel<<<(unsigned)(((size_t)MM * DI + 255) / 256), 256, 0, stream>>>(
        dbc, W_dt, b_dt, A_log, xi, dec, bx);
    scan_kernel<<<BB, 512, 0, stream>>>(ordp, parp, dec, bx);   // bx -> h
    gate_kernel<<<(unsigned)(((size_t)MM * DI + 255) / 256), 256, 0, stream>>>(
        bx, xi, zb, dbc, Dp, ubf);
    // x2 = x + u @ W_out + b_out
    gemm_wmma_kernel<1><<<dim3(256 / 128, MM / 128), 256, 0, stream>>>(
        ubf, WtOut, 512, 256, b_out, x, x2, nullptr, nullptr);

    // ---- MLP branch ----
    ln_kernel<<<MM / 8, 256, 0, stream>>>(x2, n2w, n2b, x2n, MM);
    gemm_wmma_kernel<2><<<dim3(1024 / 128, MM / 128), 256, 0, stream>>>(
        x2n, WtF1, 256, 1024, fc1_b, nullptr, nullptr, nullptr, h1);
    gemm_wmma_kernel<3><<<dim3(256 / 128, MM / 128), 256, 0, stream>>>(
        h1, WtF2, 1024, 256, fc2_b, x2, out, nullptr, nullptr);
}
